// RigelModel_4939212390966
// MI455X (gfx1250) — matrix operand order repaired and checked by reading, not compiled, not run on hardware
//
#include <hip/hip_runtime.h>
#include <hip/hip_bf16.h>

// ---- model dims ----
#define Bc    32
#define Sc    8
#define Cc    32
#define Dc    768
#define NEc   8192
#define Tc    8192
#define Hc    80
#define HOPSc 3
#define Lc    2048
#define SCc   256
#define KSPLIT 4

typedef __bf16 bf16_t;
typedef __attribute__((ext_vector_type(16))) __bf16 v16bf;
typedef __attribute__((ext_vector_type(8)))  __bf16 v8bf;
typedef __attribute__((ext_vector_type(8)))  float  v8f;

union AFrag { v16bf v; v8bf h[2]; };

__device__ __forceinline__ unsigned short f2bf(float f) {
    unsigned int u = __float_as_uint(f);
    unsigned int r = u + 0x7FFFu + ((u >> 16) & 1u);   // round-to-nearest-even
    return (unsigned short)(r >> 16);
}

// ---------------- fp32 -> bf16 elementwise convert ----------------
__global__ __launch_bounds__(256)
void k_f2bf(const float* __restrict__ src, unsigned short* __restrict__ dst, long long n) {
    long long stride = (long long)gridDim.x * blockDim.x;
    for (long long i = (long long)blockIdx.x * blockDim.x + threadIdx.x; i < n; i += stride)
        dst[i] = f2bf(src[i]);
}

// ---------------- fp32 [rows,cols] -> bf16 transpose [cols,rows] ----------------
__global__ __launch_bounds__(256)
void k_transpose_bf(const float* __restrict__ src, unsigned short* __restrict__ dst,
                    int rows, int cols) {
    __shared__ unsigned short tile[32][33];
    int c0 = blockIdx.x * 32, r0 = blockIdx.y * 32;
    int tx = threadIdx.x, ty = threadIdx.y;                  // 32 x 8
    #pragma unroll
    for (int j = 0; j < 32; j += 8)
        tile[ty + j][tx] = f2bf(src[(size_t)(r0 + ty + j) * cols + c0 + tx]);
    __syncthreads();
    #pragma unroll
    for (int j = 0; j < 32; j += 8)
        dst[(size_t)(c0 + ty + j) * rows + r0 + tx] = tile[tx][ty + j];
}

// ---------------- weighted EmbeddingBag(sum): one block per bag ----------------
__global__ __launch_bounds__(256)
void k_bag(const float* __restrict__ emb, const int* __restrict__ ids,
           const float* __restrict__ att, const int* __restrict__ offsets,
           float* __restrict__ kb) {
    int bag = blockIdx.x;                  // b*SCc + j
    int b = bag >> 8, j = bag & 255;
    int start = offsets[bag];
    int end   = (j == SCc - 1) ? Lc : offsets[bag + 1];
    int tid = threadIdx.x;
    float a0 = 0.f, a1 = 0.f, a2 = 0.f;
    for (int l = start; l < end; ++l) {
        int id  = ids[b * Lc + l];
        float w = att[b * Lc + l];
        const float* row = emb + (size_t)id * Dc;
        a0 += w * row[tid];
        a1 += w * row[tid + 256];
        a2 += w * row[tid + 512];
    }
    float* o = kb + (size_t)bag * Dc;
    o[tid] = a0; o[tid + 256] = a1; o[tid + 512] = a2;
}

// ---------------- span scores: one block per (b,s) ----------------
__global__ __launch_bounds__(256)
void k_spanscores(const float* __restrict__ se, const float* __restrict__ sw,
                  const float* __restrict__ sb, float* __restrict__ ss) {
    __shared__ float sm[256];
    int bs = blockIdx.x;
    const float* x = se + (size_t)bs * Dc;
    int tid = threadIdx.x;
    float p = x[tid] * sw[tid] + x[tid + 256] * sw[tid + 256] + x[tid + 512] * sw[tid + 512];
    sm[tid] = p; __syncthreads();
    for (int s = 128; s > 0; s >>= 1) { if (tid < s) sm[tid] += sm[tid + s]; __syncthreads(); }
    if (tid == 0) ss[bs] = sm[0] + sb[0];
}

// ---------------- sum1[bag] = dot(kb[bag], span_embs[b][j%S]) ----------------
__global__ __launch_bounds__(256)
void k_sum1(const float* __restrict__ kb, const float* __restrict__ se,
            float* __restrict__ sum1) {
    __shared__ float sm[256];
    int bag = blockIdx.x;
    int b = bag >> 8, j = bag & 255;
    const float* x = kb + (size_t)bag * Dc;
    const float* y = se + ((size_t)b * Sc + (j & (Sc - 1))) * Dc;
    int tid = threadIdx.x;
    float p = x[tid] * y[tid] + x[tid + 256] * y[tid + 256] + x[tid + 512] * y[tid + 512];
    sm[tid] = p; __syncthreads();
    for (int s = 128; s > 0; s >>= 1) { if (tid < s) sm[tid] += sm[tid + s]; __syncthreads(); }
    if (tid == 0) sum1[bag] = sm[0];
}

// ---------------- sm1 (softmax over s) then cand (softmax over 256): one block per b ----------------
__global__ __launch_bounds__(256)
void k_cand(const float* __restrict__ sum1, const float* __restrict__ ss,
            float* __restrict__ cand) {
    int b = blockIdx.x, tid = threadIdx.x;
    __shared__ float l[SCc];
    __shared__ float red[256];
    if (tid < Cc) {
        int c = tid;
        float vals[Sc];
        float m = -1e30f;
        #pragma unroll
        for (int s = 0; s < Sc; ++s) { vals[s] = sum1[b * SCc + s * Cc + c]; m = fmaxf(m, vals[s]); }
        float sum = 0.f;
        #pragma unroll
        for (int s = 0; s < Sc; ++s) { vals[s] = __expf(vals[s] - m); sum += vals[s]; }
        #pragma unroll
        for (int s = 0; s < Sc; ++s)
            l[s * Cc + c] = ss[b * Sc + s] * (vals[s] / sum);
    }
    __syncthreads();
    red[tid] = l[tid]; __syncthreads();
    for (int s = 128; s > 0; s >>= 1) { if (tid < s) red[tid] = fmaxf(red[tid], red[tid + s]); __syncthreads(); }
    float m = red[0]; __syncthreads();
    float e = __expf(l[tid] - m);
    red[tid] = e; __syncthreads();
    for (int s = 128; s > 0; s >>= 1) { if (tid < s) red[tid] += red[tid + s]; __syncthreads(); }
    cand[b * SCc + tid] = e / red[0];
}

// ---------------- deterministic scatter: ent[b][n] = sum_j (qid==n)*cand ----------------
__global__ __launch_bounds__(256)
void k_ent(const float* __restrict__ cand, const int* __restrict__ qid,
           float* __restrict__ ent) {
    int b = blockIdx.x, tid = threadIdx.x;
    __shared__ int   qs[SCc];
    __shared__ float cs[SCc];
    qs[tid] = qid[b * SCc + tid];
    cs[tid] = cand[b * SCc + tid];
    __syncthreads();
    for (int n = tid; n < NEc + 1; n += 256) {
        float s = 0.f;
        #pragma unroll 8
        for (int j = 0; j < SCc; ++j) s += (qs[j] == n) ? cs[j] : 0.f;
        ent[(size_t)b * (NEc + 1) + n] = s;
    }
}

// ---------------- softmax over NE entities: one block per b ----------------
__global__ __launch_bounds__(256)
void k_entsoftmax(const float* __restrict__ ent, float* __restrict__ x) {
    int b = blockIdx.x, tid = threadIdx.x;
    __shared__ float red[256];
    const float* e = ent + (size_t)b * (NEc + 1);
    float m = -1e30f;
    for (int n = tid; n < NEc; n += 256) m = fmaxf(m, e[n]);
    red[tid] = m; __syncthreads();
    for (int s = 128; s > 0; s >>= 1) { if (tid < s) red[tid] = fmaxf(red[tid], red[tid + s]); __syncthreads(); }
    m = red[0]; __syncthreads();
    float sum = 0.f;
    for (int n = tid; n < NEc; n += 256) sum += __expf(e[n] - m);
    red[tid] = sum; __syncthreads();
    for (int s = 128; s > 0; s >>= 1) { if (tid < s) red[tid] += red[tid + s]; __syncthreads(); }
    sum = red[0];
    for (int n = tid; n < NEc; n += 256) x[(size_t)b * NEc + n] = __expf(e[n] - m) / sum;
}

// ---------------- q = qn_emb @ red_w.T + red_b ----------------
__global__ __launch_bounds__(256)
void k_q(const float* __restrict__ qn, const float* __restrict__ rw,
         const float* __restrict__ rb, float* __restrict__ q) {
    int gid = blockIdx.x * 256 + threadIdx.x;
    if (gid >= Bc * Hc) return;
    int b = gid / Hc, h = gid % Hc;
    const float* xq = qn + (size_t)b * Dc;
    const float* wr = rw + (size_t)h * Dc;
    float s = rb[h];
    for (int d = 0; d < Dc; ++d) s += xq[d] * wr[d];
    q[gid] = s;
}

// ---------------- hop MLP: r_i = softmax(h@inf_w.T+b), att_i = h@att_w.T+b ----------------
__global__ __launch_bounds__(256)
void k_hop(const float* __restrict__ q, float* __restrict__ rhist,
           const float* __restrict__ inf_w, const float* __restrict__ inf_b,
           const float* __restrict__ att_w, const float* __restrict__ att_b,
           float* __restrict__ att, int hop) {
    int b = blockIdx.x, tid = threadIdx.x;
    __shared__ float hbuf[Hc * HOPSc];
    __shared__ float logit[Hc];
    __shared__ float red[2];
    int Hi = Hc * (hop + 1);
    if (tid < Hc) hbuf[tid] = q[b * Hc + tid];
    for (int p = 0; p < hop; ++p)
        if (tid < Hc) hbuf[(p + 1) * Hc + tid] = rhist[p * Bc * Hc + b * Hc + tid];
    __syncthreads();
    if (tid < Hc) {
        float s = inf_b[tid];
        const float* wr = inf_w + (size_t)tid * Hi;
        for (int j = 0; j < Hi; ++j) s += wr[j] * hbuf[j];
        logit[tid] = s;
    }
    __syncthreads();
    if (tid == 0) {
        float m = logit[0];
        for (int j = 1; j < Hc; ++j) m = fmaxf(m, logit[j]);
        float sum = 0.f;
        for (int j = 0; j < Hc; ++j) sum += __expf(logit[j] - m);
        red[0] = m; red[1] = sum;
        float s = att_b[0];
        for (int j = 0; j < Hi; ++j) s += att_w[j] * hbuf[j];
        att[b * HOPSc + hop] = s;
    }
    __syncthreads();
    if (tid < Hc)
        rhist[hop * Bc * Hc + b * Hc + tid] = __expf(logit[tid] - red[0]) / red[1];
}

// ---------------- v[b,t] = sum_h r[b,h]*Mp[t,h] ----------------
__global__ __launch_bounds__(256)
void k_v(const float* __restrict__ r, const float* __restrict__ Mp, float* __restrict__ v) {
    __shared__ float rl[Bc * Hc];
    int tid = threadIdx.x;
    for (int i = tid; i < Bc * Hc; i += 256) rl[i] = r[i];
    __syncthreads();
    int gid = blockIdx.x * 256 + tid;
    int b = gid & (Bc - 1);
    int t = gid >> 5;
    const float* mp = Mp + (size_t)t * Hc;
    const float* rb = rl + b * Hc;
    float s = 0.f;
    #pragma unroll 8
    for (int h = 0; h < Hc; ++h) s += rb[h] * mp[h];
    v[(size_t)b * Tc + t] = s;
}

// ---------------- WMMA bf16 GEMM:  Upart[ky][32][N] = A[32,K] x Brow[N,K]^T ----------------
// A row-major bf16 [32,K]; Brow row-major bf16 [N,K] (logical B = Brow^T, so lane K
// values are contiguous in memory).  Wave: 32x32 tile (2 M x 2 N WMMA tiles).
__global__ __launch_bounds__(128)
void k_gemm(const unsigned short* __restrict__ Au, const unsigned short* __restrict__ Bu,
            float* __restrict__ Upart, int K, int N) {
    const bf16_t* A    = (const bf16_t*)Au;
    const bf16_t* Brow = (const bf16_t*)Bu;
    const int lane = threadIdx.x & 31;
    const int wave = threadIdx.x >> 5;
    const int lmod = lane & 15;
    const int lsel = lane >> 4;                      // K-half select
    const int n0   = blockIdx.x * 128 + wave * 32;   // wave covers N [n0, n0+32)
    const int kchunk = K / KSPLIT;
    const int kbeg = blockIdx.y * kchunk;
    const int kend = kbeg + kchunk;

    v8f acc00 = {}; v8f acc01 = {}; v8f acc10 = {}; v8f acc11 = {};

    const bf16_t* a0p = A + (size_t)lmod * K;              // M tile 0 (rows 0..15)
    const bf16_t* a1p = A + (size_t)(lmod + 16) * K;       // M tile 1 (rows 16..31)
    const bf16_t* b0p = Brow + (size_t)(n0 + lmod) * K;    // N tile 0
    const bf16_t* b1p = Brow + (size_t)(n0 + 16 + lmod) * K;
    const int aoff = lsel * 8;    // A: lanes>=16 hold K 8..15 / 24..31
    const int boff = lsel * 8;    // B: lanes>=16 hold K 8..15 / 24..31

    for (int k = kbeg; k < kend; k += 32) {
        AFrag a0, a1;
        a0.h[0] = *(const v8bf*)(a0p + k + aoff);
        a0.h[1] = *(const v8bf*)(a0p + k + 16 + aoff);
        a1.h[0] = *(const v8bf*)(a1p + k + aoff);
        a1.h[1] = *(const v8bf*)(a1p + k + 16 + aoff);
        AFrag b0f, b1f;
        b0f.h[0] = *(const v8bf*)(b0p + k + boff);
        b0f.h[1] = *(const v8bf*)(b0p + k + 16 + boff);
        b1f.h[0] = *(const v8bf*)(b1p + k + boff);
        b1f.h[1] = *(const v8bf*)(b1p + k + 16 + boff);
        v16bf b0 = b0f.v;
        v16bf b1 = b1f.v;
        acc00 = __builtin_amdgcn_wmma_f32_16x16x32_bf16(false, a0.v, false, b0, (short)0, acc00, false, false);
        acc01 = __builtin_amdgcn_wmma_f32_16x16x32_bf16(false, a0.v, false, b1, (short)0, acc01, false, false);
        acc10 = __builtin_amdgcn_wmma_f32_16x16x32_bf16(false, a1.v, false, b0, (short)0, acc10, false, false);
        acc11 = __builtin_amdgcn_wmma_f32_16x16x32_bf16(false, a1.v, false, b1, (short)0, acc11, false, false);
    }

    float* out = Upart + (size_t)blockIdx.y * Bc * N;
    int col0 = n0 + lmod;
    int col1 = n0 + 16 + lmod;
    #pragma unroll
    for (int r = 0; r < 8; ++r) {
        int row0 = r + lsel * 8;          // D layout: VGPR r -> M=r (lanes<16) / M=r+8
        int row1 = 16 + r + lsel * 8;
        out[(size_t)row0 * N + col0] = acc00[r];
        out[(size_t)row0 * N + col1] = acc01[r];
        out[(size_t)row1 * N + col0] = acc10[r];
        out[(size_t)row1 * N + col1] = acc11[r];
    }
}

// ---------------- epilogue 1: w = (sum_p Upart) * v  -> bf16 ----------------
__global__ __launch_bounds__(256)
void k_epi_mulv(const float* __restrict__ up, const float* __restrict__ v,
                unsigned short* __restrict__ wbf) {
    int gid = blockIdx.x * 256 + threadIdx.x;   // over Bc*Tc
    float s = 0.f;
    #pragma unroll
    for (int p = 0; p < KSPLIT; ++p) s += up[(size_t)p * Bc * Tc + gid];
    wbf[gid] = f2bf(s * v[gid]);
}

// ---------------- epilogue 2: x_i = sum_p Upart ----------------
__global__ __launch_bounds__(256)
void k_epi_sum(const float* __restrict__ up, float* __restrict__ xo) {
    int gid = blockIdx.x * 256 + threadIdx.x;   // over Bc*NEc
    float s = 0.f;
    #pragma unroll
    for (int p = 0; p < KSPLIT; ++p) s += up[(size_t)p * Bc * NEc + gid];
    xo[gid] = s;
}

// ---------------- final: sigmoid( sum_i softmax(att)[i] * x_i ) ----------------
__global__ __launch_bounds__(256)
void k_final(const float* __restrict__ xh, const float* __restrict__ att,
             float* __restrict__ out) {
    int gid = blockIdx.x * 256 + threadIdx.x;   // over Bc*NEc
    int b = gid >> 13;
    float a0 = att[b * 3 + 0], a1 = att[b * 3 + 1], a2 = att[b * 3 + 2];
    float m = fmaxf(a0, fmaxf(a1, a2));
    float e0 = __expf(a0 - m), e1 = __expf(a1 - m), e2 = __expf(a2 - m);
    float inv = 1.f / (e0 + e1 + e2);
    float z = (e0 * xh[gid]
             + e1 * xh[(size_t)Bc * NEc + gid]
             + e2 * xh[2 * (size_t)Bc * NEc + gid]) * inv;
    out[gid] = 1.f / (1.f + __expf(-z));
}

extern "C" void kernel_launch(void* const* d_in, const int* in_sizes, int n_in,
                              void* d_out, int out_size, void* d_ws, size_t ws_size,
                              hipStream_t stream) {
    // inputs (setup_inputs dict order)
    const float* span_embs = (const float*)d_in[0];
    const float* att_tr    = (const float*)d_in[1];
    const float* qn_emb    = (const float*)d_in[2];
    const int*   trip_ids  = (const int*)d_in[3];
    const int*   qid_inds  = (const int*)d_in[4];
    const int*   offsets   = (const int*)d_in[5];
    const float* emb_table = (const float*)d_in[6];
    const float* span_w    = (const float*)d_in[7];
    const float* span_b    = (const float*)d_in[8];
    const float* red_w     = (const float*)d_in[9];
    const float* red_b     = (const float*)d_in[10];
    const float* Ms        = (const float*)d_in[11];
    const float* Mo        = (const float*)d_in[12];
    const float* Mp        = (const float*)d_in[13];
    float* out = (float*)d_out;

    // workspace carve-up (~302 MB)
    char* w = (char*)d_ws;
    auto alloc = [&](size_t bytes) -> void* {
        void* p = (void*)w;
        w += (bytes + 255) & ~(size_t)255;
        return p;
    };
    unsigned short* ms_bf  = (unsigned short*)alloc((size_t)Tc * NEc * 2);   // 134 MB
    unsigned short* mot_bf = (unsigned short*)alloc((size_t)NEc * Tc * 2);   // 134 MB
    float* upart = (float*)alloc((size_t)KSPLIT * Bc * Tc * 4);              // 4 MB
    float* kb    = (float*)alloc((size_t)Bc * SCc * Dc * 4);                 // 25 MB
    float* ent   = (float*)alloc((size_t)Bc * (NEc + 1) * 4);
    float* x0    = (float*)alloc((size_t)Bc * NEc * 4);
    float* xh    = (float*)alloc((size_t)HOPSc * Bc * NEc * 4);
    float* vbuf  = (float*)alloc((size_t)Bc * Tc * 4);
    unsigned short* xbf = (unsigned short*)alloc((size_t)Bc * NEc * 2);
    unsigned short* wbf = (unsigned short*)alloc((size_t)Bc * Tc * 2);
    float* ss    = (float*)alloc((size_t)Bc * Sc * 4);
    float* sum1  = (float*)alloc((size_t)Bc * SCc * 4);
    float* cand  = (float*)alloc((size_t)Bc * SCc * 4);
    float* qbuf  = (float*)alloc((size_t)Bc * Hc * 4);
    float* rhist = (float*)alloc((size_t)HOPSc * Bc * Hc * 4);
    float* attb  = (float*)alloc((size_t)Bc * HOPSc * 4);
    (void)ws_size; (void)n_in; (void)in_sizes; (void)out_size;

    // one-time (per call) bf16 prep of the big matrices
    k_f2bf<<<32768, 256, 0, stream>>>(Ms, ms_bf, (long long)Tc * NEc);
    {
        dim3 g(NEc / 32, Tc / 32), b(32, 8);
        k_transpose_bf<<<g, b, 0, stream>>>(Mo, mot_bf, Tc, NEc);
    }

    // ---- EntityResolution ----
    k_bag<<<Bc * SCc, 256, 0, stream>>>(emb_table, trip_ids, att_tr, offsets, kb);
    k_spanscores<<<Bc * Sc, 256, 0, stream>>>(span_embs, span_w, span_b, ss);
    k_sum1<<<Bc * SCc, 256, 0, stream>>>(kb, span_embs, sum1);
    k_cand<<<Bc, 256, 0, stream>>>(sum1, ss, cand);
    k_ent<<<Bc, 256, 0, stream>>>(cand, qid_inds, ent);
    k_entsoftmax<<<Bc, 256, 0, stream>>>(ent, x0);

    // ---- InferenceModule ----
    k_q<<<(Bc * Hc + 255) / 256, 256, 0, stream>>>(qn_emb, red_w, red_b, qbuf);

    const float* xcur = x0;
    for (int i = 0; i < HOPSc; ++i) {
        const float* inf_w = (const float*)d_in[14 + 4 * i];
        const float* inf_b = (const float*)d_in[15 + 4 * i];
        const float* att_w = (const float*)d_in[16 + 4 * i];
        const float* att_b = (const float*)d_in[17 + 4 * i];

        k_hop<<<Bc, 256, 0, stream>>>(qbuf, rhist, inf_w, inf_b, att_w, att_b, attb, i);
        k_v<<<(Bc * Tc) / 256, 256, 0, stream>>>(rhist + (size_t)i * Bc * Hc, Mp, vbuf);
        k_f2bf<<<256, 256, 0, stream>>>(xcur, xbf, (long long)Bc * NEc);

        {   // U = x @ Ms^T   (A = xbf [32,NE], Brow = Ms_bf [T,NE])
            dim3 g(Tc / 128, KSPLIT);
            k_gemm<<<g, 128, 0, stream>>>(xbf, ms_bf, upart, NEc, Tc);
        }
        k_epi_mulv<<<(Bc * Tc) / 256, 256, 0, stream>>>(upart, vbuf, wbf);
        {   // Xnew = w @ Mo   (A = wbf [32,T], Brow = Mo^T bf16 [NE,T])
            dim3 g(NEc / 128, KSPLIT);
            k_gemm<<<g, 128, 0, stream>>>(wbf, mot_bf, upart, Tc, NEc);
        }
        float* xo = xh + (size_t)i * Bc * NEc;
        k_epi_sum<<<(Bc * NEc) / 256, 256, 0, stream>>>(upart, xo);
        xcur = xo;
    }

    k_final<<<(Bc * NEc) / 256, 256, 0, stream>>>(xh, attb, out);
}
